// DynamicDownsampling_41394894799614
// MI455X (gfx1250) — matrix operand order repaired and checked by reading, not compiled, not run on hardware
//
#include <hip/hip_runtime.h>

// ---------------------------------------------------------------------------
// DynamicDownsampling for MI455X (gfx1250, wave32, WMMA).
// Heavy 3x3 64->64 convs as implicit GEMM via v_wmma_f32_16x16x32_bf16.
// ---------------------------------------------------------------------------

typedef __attribute__((ext_vector_type(16))) __bf16 bf16x16;
typedef __attribute__((ext_vector_type(8)))  __bf16 bf16x8;
typedef __attribute__((ext_vector_type(8)))  float  f32x8;

#define CH    64
#define HH    128
#define WWD   128
#define HWSZ  (HH*WWD)           // 16384
#define FRN   16                 // B*T frames
#define FSZ   (FRN*CH*HWSZ)      // 16,777,216 floats per tensor
#define NEGSL 0.2f

union AccU { f32x8 v; float f[8]; };

__device__ __forceinline__ f32x8 wmma_bf16(bf16x16 a, bf16x16 b, f32x8 c) {
  // D = A(16x32 bf16) * B(32x16 bf16) + C(16x16 f32)
  return __builtin_amdgcn_wmma_f32_16x16x32_bf16(false, a, false, b,
                                                 (short)0, c, false, false);
}

// ---------------------------------------------------------------------------
// Weight pre-pack: (O=64, C=64, 3, 3) f32 -> per-lane WMMA B-fragment bf16.
// frag[((r*2+kc)*4 + n)*512 + lane*16 + e] with
//   K = (e<8 ? e : e+8) + (lane>=16 ? 8:0),  N = lane%16,  ic = kc*32+K, oc = n*16+N.
// ---------------------------------------------------------------------------
__global__ void pack_w3_kernel(const float* __restrict__ w, __bf16* __restrict__ frag) {
  int i = blockIdx.x * 256 + threadIdx.x;          // 9*2*4*512 = 36864
  if (i >= 36864) return;
  int e    = i & 15;
  int lane = (i >> 4) & 31;
  int grp  = i >> 9;            // (r*2+kc)*4 + n
  int n    = grp & 3;
  int kc   = (grp >> 2) & 1;
  int r    = grp >> 3;          // tap = dy*3+dx
  int k    = ((lane >= 16) ? 8 : 0) + (e < 8 ? e : e + 8);
  int oc   = n * 16 + (lane & 15);
  int ic   = kc * 32 + k;
  frag[i] = (__bf16)w[(oc * 64 + ic) * 9 + r];
}

__global__ void pack_w1_kernel(const float* __restrict__ w, __bf16* __restrict__ frag) {
  int i = blockIdx.x * 256 + threadIdx.x;          // 2*4*512 = 4096
  if (i >= 4096) return;
  int e    = i & 15;
  int lane = (i >> 4) & 31;
  int grp  = i >> 9;            // kc*4 + n
  int n    = grp & 3;
  int kc   = grp >> 2;
  int k    = ((lane >= 16) ? 8 : 0) + (e < 8 ? e : e + 8);
  int oc   = n * 16 + (lane & 15);
  int ic   = kc * 32 + k;
  frag[i] = (__bf16)w[oc * 64 + ic];
}

// ---------------------------------------------------------------------------
// 3x3 conv, 64->64 ch, pad 1, via WMMA implicit GEMM.
// grid = (2 col-tiles, 128 rows, 16 frames), block = 128 (4 waves).
// Each wave: 16 positions x 64 out-ch (4 accumulators), K = 9 taps * 2 chunks.
// in_is_x: input uses original (B,C,T,H,W) strides. resid (x-layout) optional.
// ---------------------------------------------------------------------------
__global__ __launch_bounds__(128)
void conv3x3_wmma_kernel(const float* __restrict__ in, int in_is_x,
                         const __bf16* __restrict__ wfrag,
                         const float* __restrict__ bias,
                         const float* __restrict__ resid,
                         float* __restrict__ out, int act)
{
  __shared__ __attribute__((aligned(16))) __bf16 tile[3 * 66 * 64];

  const int tid = threadIdx.x;
  const int f   = blockIdx.z;
  const int h   = blockIdx.y;
  const int w0  = blockIdx.x * 64;
  const int bb  = f >> 3, tt = f & 7;

  long inBase, cStride;
  if (in_is_x) { inBase = ((long)bb * 512 + tt) * HWSZ; cStride = 8L * HWSZ; }
  else         { inBase = (long)f * CH * HWSZ;          cStride = HWSZ;      }

  // Stage 3 rows x 64 ch x 66 cols (halo) as bf16, channel-contiguous.
  for (int p = tid; p < 3 * CH; p += 128) {
    int row = p >> 6;
    int c   = p & 63;
    int gh  = h + row - 1;
    __bf16* dst = &tile[(row * 66) * 64 + c];
    if (gh < 0 || gh >= HH) {
      for (int col = 0; col < 66; ++col) dst[col * 64] = (__bf16)0.0f;
    } else {
      const float* src = in + inBase + (long)c * cStride + (long)gh * WWD;
      __builtin_prefetch(src + w0, 0, 0);   // global_prefetch_b8
      for (int col = 0; col < 66; ++col) {
        int gw = w0 + col - 1;
        float v = (gw >= 0 && gw < WWD) ? src[gw] : 0.0f;
        dst[col * 64] = (__bf16)v;
      }
    }
  }
  __syncthreads();

  const int wave = tid >> 5;
  const int lane = tid & 31;
  const int lm   = lane & 15;
  const int hi   = lane >> 4;

  AccU acc[4];
  #pragma unroll
  for (int n = 0; n < 4; ++n) {
    float bv = bias[n * 16 + lm];            // C/D layout: N = lane%16
    #pragma unroll
    for (int v = 0; v < 8; ++v) acc[n].f[v] = bv;
  }

  #pragma unroll
  for (int r = 0; r < 9; ++r) {
    const int dy = r / 3, dx = r % 3;
    const int colA = wave * 16 + lm + dx;    // 0..65 (halo coords)
    #pragma unroll
    for (int kc = 0; kc < 2; ++kc) {
      // A fragment: lane holds M=lm; e<8 -> K=e(+8*hi), e>=8 -> K=e+8(+8*hi):
      // two runs of 8 consecutive channels => two ds_load_b128.
      const __bf16* ap = &tile[(dy * 66 + colA) * 64 + kc * 32 + hi * 8];
      bf16x8 alo = *(const bf16x8*)ap;
      bf16x8 ahi = *(const bf16x8*)(ap + 16);
      bf16x16 a = __builtin_shufflevector(alo, ahi,
                    0,1,2,3,4,5,6,7,8,9,10,11,12,13,14,15);
      const __bf16* wb = wfrag + (long)((r * 2 + kc) * 4) * 512 + lane * 16;
      #pragma unroll
      for (int n = 0; n < 4; ++n) {
        bf16x16 bm = *(const bf16x16*)(wb + n * 512);
        acc[n].v = wmma_bf16(a, bm, acc[n].v);
      }
    }
  }

  // Epilogue: optional LeakyReLU, optional residual (x-layout), store f32.
  const int wp = w0 + wave * 16;
  #pragma unroll
  for (int n = 0; n < 4; ++n) {
    int  ch    = n * 16 + lm;
    long obase = ((long)f * CH + ch) * HWSZ + (long)h * WWD;
    long rbase = 0;
    if (resid) rbase = ((long)bb * 512 + tt + (long)ch * 8) * HWSZ + (long)h * WWD;
    #pragma unroll
    for (int v = 0; v < 8; ++v) {
      int   m   = v + hi * 8;                // C/D layout: M = v (+8 for hi half)
      float val = acc[n].f[v];
      if (act)   val = (val >= 0.0f) ? val : NEGSL * val;
      if (resid) val += resid[rbase + wp + m];
      out[obase + wp + m] = val;
    }
  }
}

// ---------------------------------------------------------------------------
// 1x1 conv (64x64 GEMM) via WMMA. grid = (2, 128, 16), block = 128.
// ---------------------------------------------------------------------------
__global__ __launch_bounds__(128)
void conv1x1_wmma_kernel(const float* __restrict__ in,
                         const __bf16* __restrict__ wfrag,
                         const float* __restrict__ bias,
                         float* __restrict__ out)
{
  __shared__ __attribute__((aligned(16))) __bf16 tile[64 * 64];
  const int tid = threadIdx.x;
  const int f = blockIdx.z, h = blockIdx.y, w0 = blockIdx.x * 64;

  {
    int c    = tid & 63;
    int half = tid >> 6;
    const float* src = in + ((long)f * CH + c) * HWSZ + (long)h * WWD + w0 + half * 32;
    __bf16* dst = &tile[(half * 32) * 64 + c];
    for (int col = 0; col < 32; ++col) dst[col * 64] = (__bf16)src[col];
  }
  __syncthreads();

  const int wave = tid >> 5, lane = tid & 31, lm = lane & 15, hi = lane >> 4;
  AccU acc[4];
  #pragma unroll
  for (int n = 0; n < 4; ++n) {
    float bv = bias[n * 16 + lm];
    #pragma unroll
    for (int v = 0; v < 8; ++v) acc[n].f[v] = bv;
  }
  #pragma unroll
  for (int kc = 0; kc < 2; ++kc) {
    const __bf16* ap = &tile[(wave * 16 + lm) * 64 + kc * 32 + hi * 8];
    bf16x8 alo = *(const bf16x8*)ap;
    bf16x8 ahi = *(const bf16x8*)(ap + 16);
    bf16x16 a = __builtin_shufflevector(alo, ahi,
                  0,1,2,3,4,5,6,7,8,9,10,11,12,13,14,15);
    const __bf16* wb = wfrag + (long)(kc * 4) * 512 + lane * 16;
    #pragma unroll
    for (int n = 0; n < 4; ++n) {
      bf16x16 bm = *(const bf16x16*)(wb + n * 512);
      acc[n].v = wmma_bf16(a, bm, acc[n].v);
    }
  }
  const int wp = w0 + wave * 16;
  #pragma unroll
  for (int n = 0; n < 4; ++n) {
    int  ch    = n * 16 + lm;
    long obase = ((long)f * CH + ch) * HWSZ + (long)h * WWD;
    #pragma unroll
    for (int v = 0; v < 8; ++v)
      out[obase + wp + v + hi * 8] = acc[n].f[v];
  }
}

// ---------------------------------------------------------------------------
// 4-direction grouped spatial shift (s=4, zero fill) * gate.
// ---------------------------------------------------------------------------
__global__ void shiftmul_kernel(const float* __restrict__ r, float* __restrict__ o) {
  long i = (long)blockIdx.x * 256 + threadIdx.x;
  if (i >= (long)FSZ) return;
  int w  = (int)(i & 127);
  int hh = (int)((i >> 7) & 127);
  int c  = (int)((i >> 14) & 63);
  int g  = c >> 4;                       // 16 channels per group
  float s = 0.0f;
  if      (g == 0) { if (w  >= 4)       s = r[i - 4];        }  // shift right
  else if (g == 1) { if (w  <  WWD - 4) s = r[i + 4];        }  // shift left
  else if (g == 2) { if (hh >= 4)       s = r[i - 4 * WWD];  }  // shift down
  else             { if (hh <  HH - 4)  s = r[i + 4 * WWD];  }  // shift up
  o[i] = s * r[i];
}

// ---------------------------------------------------------------------------
// 3x3 conv 64->9 ch, stride 2, pad 1 (zero). Tiny (0.7 GFLOP) -> VALU.
// out layout: [f][9][64][64] flat == thread index.
// ---------------------------------------------------------------------------
__global__ void convs2_kernel(const float* __restrict__ in, const float* __restrict__ wt,
                              const float* __restrict__ bias, float* __restrict__ out) {
  int i = blockIdx.x * 256 + threadIdx.x;          // 16*9*64*64 = 589824
  if (i >= FRN * 9 * 64 * 64) return;
  int wo = i & 63;
  int ho = (i >> 6) & 63;
  int oc = (i >> 12) % 9;
  int f  = i / (4096 * 9);
  float s = bias[oc];
  for (int ic = 0; ic < 64; ++ic) {
    const float* ip = in + ((long)f * CH + ic) * HWSZ;
    const float* wp = wt + (long)(oc * 64 + ic) * 9;
    #pragma unroll
    for (int ky = 0; ky < 3; ++ky) {
      int hy = 2 * ho + ky - 1;
      if (hy < 0 || hy >= HH) continue;
      #pragma unroll
      for (int kx = 0; kx < 3; ++kx) {
        int wx = 2 * wo + kx - 1;
        if (wx < 0 || wx >= WWD) continue;
        s += wp[ky * 3 + kx] * ip[hy * WWD + wx];
      }
    }
  }
  out[i] = s;
}

// ---------------------------------------------------------------------------
// Fused softmax(72 = T*K*K) + replicate-padded strided patch reduction.
// block = 64 threads (one per channel), blockIdx = b*4096 + ho*64 + wo.
// ---------------------------------------------------------------------------
__global__ __launch_bounds__(64)
void softmax_reduce_kernel(const float* __restrict__ x5, const float* __restrict__ kf2,
                           float* __restrict__ out) {
  __shared__ float sm[72];
  int blk = blockIdx.x;
  int wo  = blk & 63;
  int ho  = (blk >> 6) & 63;
  int b   = blk >> 12;
  int tid = threadIdx.x;
  for (int j = tid; j < 72; j += 64) {
    int t = j / 9, kk = j % 9;
    sm[j] = kf2[(((long)(b * 8 + t) * 9 + kk) << 12) + ho * 64 + wo];
  }
  __syncthreads();
  float mx = -1e30f;
  for (int j = 0; j < 72; ++j) mx = fmaxf(mx, sm[j]);
  float den = 0.0f;
  for (int j = 0; j < 72; ++j) den += __expf(sm[j] - mx);

  int c = tid;
  float s = 0.0f;
  for (int t = 0; t < 8; ++t) {
    const float* xp = x5 + ((long)(b * 8 + t) * CH + c) * HWSZ;
    #pragma unroll
    for (int ki = 0; ki < 3; ++ki) {
      int hsrc = 2 * ho + ki - 1;
      hsrc = min(max(hsrc, 0), HH - 1);           // replicate padding
      #pragma unroll
      for (int kj = 0; kj < 3; ++kj) {
        int wsrc = 2 * wo + kj - 1;
        wsrc = min(max(wsrc, 0), WWD - 1);
        s += __expf(sm[t * 9 + ki * 3 + kj] - mx) * xp[hsrc * WWD + wsrc];
      }
    }
  }
  out[(((long)(b * CH + c)) << 12) + ho * 64 + wo] = s / den;
}

// ---------------------------------------------------------------------------
// Host side
// ---------------------------------------------------------------------------
extern "C" void kernel_launch(void* const* d_in, const int* in_sizes, int n_in,
                              void* d_out, int out_size, void* d_ws, size_t ws_size,
                              hipStream_t stream) {
  (void)in_sizes; (void)n_in; (void)out_size; (void)ws_size;
  const float* x   = (const float*)d_in[0];
  const float* w1  = (const float*)d_in[1];
  const float* b1  = (const float*)d_in[2];
  const float* w2  = (const float*)d_in[3];
  const float* b2  = (const float*)d_in[4];
  const float* wk1 = (const float*)d_in[5];
  const float* bk1 = (const float*)d_in[6];
  const float* wk2 = (const float*)d_in[7];
  const float* bk2 = (const float*)d_in[8];
  const float* wk3 = (const float*)d_in[9];
  const float* bk3 = (const float*)d_in[10];
  float* out = (float*)d_out;

  float* ws   = (float*)d_ws;
  float* bufA = ws;                        // res1, later k
  float* bufB = ws + (size_t)FSZ;          // res_s, later kf
  float* bufC = ws + 2 * (size_t)FSZ;      // x5 (persists to the end)
  float* kf2  = ws + 3 * (size_t)FSZ;      // 16*9*64*64
  __bf16* frag = (__bf16*)(ws + 3 * (size_t)FSZ + 589824);
  __bf16* fW1  = frag;
  __bf16* fW2  = frag + 36864;
  __bf16* fWk2 = frag + 2 * 36864;
  __bf16* fWk1 = frag + 3 * 36864;

  // Weight fragment packing (tiny)
  pack_w3_kernel<<<144, 256, 0, stream>>>(w1,  fW1);
  pack_w3_kernel<<<144, 256, 0, stream>>>(w2,  fW2);
  pack_w3_kernel<<<144, 256, 0, stream>>>(wk2, fWk2);
  pack_w1_kernel<<<16,  256, 0, stream>>>(wk1, fWk1);

  dim3 cgrid(2, 128, 16);   // col-tiles x rows x frames

  // res1 = lrelu(conv1(x))                  (x strided input)
  conv3x3_wmma_kernel<<<cgrid, 128, 0, stream>>>(x, 1, fW1, b1, nullptr, bufA, 1);
  // res_s = shift(res1) * res1
  shiftmul_kernel<<<FSZ / 256, 256, 0, stream>>>(bufA, bufB);
  // x5 = conv2(res_s) + x                   (residual read from x strides)
  conv3x3_wmma_kernel<<<cgrid, 128, 0, stream>>>(bufB, 0, fW2, b2, x, bufC, 0);
  // k = wk1 . x5 + bk1                      (1x1 GEMM)
  conv1x1_wmma_kernel<<<cgrid, 128, 0, stream>>>(bufC, fWk1, bk1, bufA);
  // kf = lrelu(conv_wk2(k))
  conv3x3_wmma_kernel<<<cgrid, 128, 0, stream>>>(bufA, 0, fWk2, bk2, nullptr, bufB, 1);
  // kf2 = conv_wk3(kf), stride 2, 9 ch
  convs2_kernel<<<2304, 256, 0, stream>>>(bufB, wk3, bk3, kf2);
  // out = softmax-weighted patch reduce of x5
  softmax_reduce_kernel<<<2 * 64 * 64, 64, 0, stream>>>(bufC, kf2, out);
}
